// CTCFBlock_64089501991245
// MI455X (gfx1250) — matrix operand order repaired and checked by reading, not compiled, not run on hardware
//
#include <hip/hip_runtime.h>
#include <math.h>

#define DEV __device__ __forceinline__

typedef __attribute__((ext_vector_type(16))) __bf16 v16bf;
typedef __attribute__((ext_vector_type(8)))  float  v8f;

union FragBF { v16bf v; unsigned u[8]; };
union FragF  { v8f   v; float    f[8]; };

// ---------- scalar helpers ----------
DEV unsigned short f2bf(float f){
  unsigned u = __float_as_uint(f);
  u += 0x7FFFu + ((u >> 16) & 1u);     // round-to-nearest-even
  return (unsigned short)(u >> 16);
}

DEV v8f vzero8(){
  FragF z;
#pragma unroll
  for (int i = 0; i < 8; ++i) z.f[i] = 0.f;
  return z.v;
}

DEV v8f wmma_bf16(v16bf a, v16bf b, v8f c){
  return __builtin_amdgcn_wmma_f32_16x16x32_bf16(false, a, false, b, (short)0, c, false, false);
}

// ---------- CDNA5 async global->LDS (ASYNCcnt path) ----------
// GLOBAL_LOAD_ASYNC_TO_LDS_B128: VDST = VGPR with LDS byte address
// (offset from the LDS base); 64-bit global address in VGPR pair.
// Generic LDS pointers carry the LDS offset in their low 32 bits (ISA 10.2).
DEV void async_ld_b128(const void* gaddr, const void* lds_dst){
  unsigned l = (unsigned)(size_t)lds_dst;
  asm volatile("global_load_async_to_lds_b128 %0, %1, off"
               :: "v"(l), "v"(gaddr) : "memory");
}
DEV void wait_async0(){
#if __has_builtin(__builtin_amdgcn_s_wait_asynccnt)
  __builtin_amdgcn_s_wait_asynccnt(0);
#else
  asm volatile("s_wait_asynccnt 0" ::: "memory");
#endif
}

// ---------- fragment loaders (16-bit WMMA operands, contiguous k order, wave32) ----------
// A 16x32: lane L: row = L%16, half = L/16; vgpr v holds K pair
//   k = 2v + 16*half  (pairs contiguous along K)
DEV v16bf fragA_load(const unsigned short* p, int ld){
  FragBF f; int L = threadIdx.x & 31; int r = L & 15; int hf = L >> 4;
#pragma unroll
  for (int v = 0; v < 8; ++v){
    int k = 2 * v + 16 * hf;
    f.u[v] = *(const unsigned*)(p + r * ld + k);
  }
  return f.v;
}
// A 16x32 with K padded: only k<16 valid (hd=16 attention)
DEV v16bf fragA_load_pad16(const unsigned short* p, int ld){
  FragBF f; int L = threadIdx.x & 31; int r = L & 15; int hf = L >> 4;
#pragma unroll
  for (int v = 0; v < 8; ++v){
    int k = 2 * v + 16 * hf;
    f.u[v] = (k < 16) ? *(const unsigned*)(p + r * ld + k) : 0u;
  }
  return f.v;
}
// B 32x16, source stored n-major ("B^T row-major" [n][k]):
//   lane L: col n = L%16, half = L/16; vgpr v holds k = {2v,2v+1} + 16*half
DEV v16bf fragB_load_nmajor(const unsigned short* p, int ld){
  FragBF f; int L = threadIdx.x & 31; int n = L & 15; int hf = L >> 4;
#pragma unroll
  for (int v = 0; v < 8; ++v){
    int k = 2 * v + 16 * hf;
    f.u[v] = *(const unsigned*)(p + n * ld + k);
  }
  return f.v;
}
// B = K^T built from token-major K rows [token][hd]; only k<16 valid
DEV v16bf fragB_load_kt_pad16(const unsigned short* p, int ld){
  FragBF f; int L = threadIdx.x & 31; int n = L & 15; int hf = L >> 4;
#pragma unroll
  for (int v = 0; v < 8; ++v){
    int k = 2 * v + 16 * hf;
    f.u[v] = (k < 16) ? *(const unsigned*)(p + n * ld + k) : 0u;
  }
  return f.v;
}
// B 32x16 from k-major storage [k][n] (V matrix): gather & pack
DEV v16bf fragB_load_kmajor(const unsigned short* p, int ld){
  FragBF f; int L = threadIdx.x & 31; int n = L & 15; int hf = L >> 4;
#pragma unroll
  for (int v = 0; v < 8; ++v){
    int k = 2 * v + 16 * hf;
    unsigned lo = p[k * ld + n];
    unsigned hi = p[(k + 1) * ld + n];
    f.u[v] = lo | (hi << 16);
  }
  return f.v;
}

// ===================== prep kernels =====================
// x [B][64][32768] f32 -> xT [B][32768][64] bf16
__global__ void k_xpose(const float* __restrict__ x, unsigned short* __restrict__ xT){
  unsigned idx = blockIdx.x * 256u + threadIdx.x;
  if (idx >= 2u * 32768u * 64u) return;
  int ci = idx & 63; int v = (idx >> 6) & 32767; int b = idx >> 21;
  xT[idx] = f2bf(x[((size_t)b * 64 + ci) * 32768 + v]);
}

// conv weight [CO=128][CIN][27] f32 -> [27][128][CIN] bf16
__global__ void k_wprep_conv(const float* __restrict__ w, unsigned short* __restrict__ wt, int CIN){
  int idx = blockIdx.x * 256 + threadIdx.x;
  int total = 27 * 128 * CIN;
  if (idx >= total) return;
  int ci = idx % CIN; int co = (idx / CIN) % 128; int t = idx / (CIN * 128);
  wt[idx] = f2bf(w[((size_t)co * CIN + ci) * 27 + t]);
}

// dense weight -> Bt [N][K] bf16.  trans: in is [K][N]; else in already [N][K]
__global__ void k_wprep_mat(const float* __restrict__ w, unsigned short* __restrict__ wt,
                            int K, int N, int trans){
  int idx = blockIdx.x * 256 + threadIdx.x;
  if (idx >= K * N) return;
  int k = idx % K, n = idx / K;
  wt[idx] = f2bf(trans ? w[(size_t)k * N + n] : w[idx]);
}

// ===================== implicit-GEMM 3x3x3 conv =====================
// A [B][32768][CIN] bf16 voxel-major, Wt [27][128][CIN] bf16 -> out [B][32768][128] f32
// Double-buffered: async tile k+1 issues while WMMA consumes tile k.
__global__ void __launch_bounds__(256) k_conv3(const unsigned short* __restrict__ Ain,
                                               const unsigned short* __restrict__ Wt,
                                               float* __restrict__ out, int CIN){
  __shared__ unsigned short lds_a[2][64 * 40];
  __shared__ unsigned short lds_b[2][128 * 40];
  int tid = threadIdx.x; int wid = tid >> 5; int lane = tid & 31;
  int m0 = blockIdx.x * 64; int b = m0 >> 15; int vb = m0 & 32767;
  int msub = (wid & 3) * 16, nsub = (wid >> 2) * 64;

  FragF acc[4];
#pragma unroll
  for (int j = 0; j < 4; ++j) acc[j].v = vzero8();

  int arow = tid >> 2, ac8 = (tid & 3) * 8;
  int av = vb + arow;
  int aw = av & 31, ah = (av >> 5) & 31, ad = av >> 10;
  int brow = tid >> 1, bc16 = (tid & 1) * 16;
  int kcs = (CIN == 64) ? 1 : 2;        // log2(k-chunks per tap); chunks = CIN/32
  int nk = 27 << kcs;

  auto issue = [&](int step, int buf){
    int tap = step >> kcs;
    int kc  = step & ((1 << kcs) - 1);
    int kd = tap / 9 - 1, khh = (tap / 3) % 3 - 1, kww = tap % 3 - 1;
    int zd = ad + kd, zh = ah + khh, zw = aw + kww;
    bool inb = ((unsigned)zd < 32u) && ((unsigned)zh < 32u) && ((unsigned)zw < 32u);
    unsigned short* la = &lds_a[buf][arow * 40 + ac8];
    if (inb){
      int zv = ((zd << 5) + zh) * 32 + zw;
      async_ld_b128(Ain + ((size_t)b * 32768 + zv) * CIN + kc * 32 + ac8, la);
    } else {
      unsigned* d = (unsigned*)la;
      d[0] = 0u; d[1] = 0u; d[2] = 0u; d[3] = 0u;   // zero-pad halo
    }
    const unsigned short* gb = Wt + ((size_t)tap * 128 + brow) * CIN + kc * 32 + bc16;
    unsigned short* lb = &lds_b[buf][brow * 40 + bc16];
    async_ld_b128(gb, lb);
    async_ld_b128(gb + 8, lb + 8);
  };

  issue(0, 0);
  for (int ks = 0; ks < nk; ++ks){
    int cur = ks & 1;
    wait_async0();
    __syncthreads();
    if (ks + 1 < nk) issue(ks + 1, 1 - cur);
    v16bf fa = fragA_load(&lds_a[cur][msub * 40], 40);
#pragma unroll
    for (int j = 0; j < 4; ++j){
      v16bf fb = fragB_load_nmajor(&lds_b[cur][(nsub + 16 * j) * 40], 40);
      acc[j].v = wmma_bf16(fa, fb, acc[j].v);
    }
    __syncthreads();
  }

  int hf = lane >> 4, nl = lane & 15;
#pragma unroll
  for (int j = 0; j < 4; ++j){
    int n = nsub + 16 * j + nl;
#pragma unroll
    for (int v = 0; v < 8; ++v){
      int m = msub + v + 8 * hf;
      out[((size_t)(m0 + m)) * 128 + n] = acc[j].f[v];
    }
  }
}

// ===================== generic GEMM =====================
// A [M][K] bf16, Bt [N][K] bf16; epilogue: +bias, act(1=GELU exact), +resid; f32/bf16 out
__global__ void __launch_bounds__(256) k_gemm(const unsigned short* __restrict__ A,
                                              const unsigned short* __restrict__ Bt,
                                              const float* __restrict__ bias,
                                              const float* __restrict__ resid,
                                              float* __restrict__ outF,
                                              unsigned short* __restrict__ outB,
                                              int M, int N, int K, int act){
  __shared__ unsigned short lds_a[2][64 * 40];
  __shared__ unsigned short lds_b[2][64 * 40];
  int tid = threadIdx.x, wid = tid >> 5, lane = tid & 31;
  int m0 = blockIdx.x * 64, n0 = blockIdx.y * 64;
  int msub = (wid & 3) * 16, nsub = (wid >> 2) * 32;
  FragF acc[2];
  acc[0].v = vzero8(); acc[1].v = vzero8();
  int arow = tid >> 2, ac8 = (tid & 3) * 8;

  const unsigned short* gA = A  + (size_t)(m0 + arow) * K + ac8;
  const unsigned short* gB = Bt + (size_t)(n0 + arow) * K + ac8;
  int nk = K >> 5;

  auto issue = [&](int ks, int buf){
    async_ld_b128(gA + ks * 32, &lds_a[buf][arow * 40 + ac8]);
    async_ld_b128(gB + ks * 32, &lds_b[buf][arow * 40 + ac8]);
  };

  issue(0, 0);
  for (int ks = 0; ks < nk; ++ks){
    int cur = ks & 1;
    wait_async0();
    __syncthreads();
    if (ks + 1 < nk) issue(ks + 1, 1 - cur);
    v16bf fa = fragA_load(&lds_a[cur][msub * 40], 40);
#pragma unroll
    for (int j = 0; j < 2; ++j){
      v16bf fb = fragB_load_nmajor(&lds_b[cur][(nsub + 16 * j) * 40], 40);
      acc[j].v = wmma_bf16(fa, fb, acc[j].v);
    }
    __syncthreads();
  }

  int hf = lane >> 4, nl = lane & 15;
#pragma unroll
  for (int j = 0; j < 2; ++j){
    int n = n0 + nsub + 16 * j + nl;
    float bv = bias ? bias[n] : 0.f;
#pragma unroll
    for (int v = 0; v < 8; ++v){
      int m = m0 + msub + v + 8 * hf;
      float x = acc[j].f[v] + bv;
      if (act == 1) x = 0.5f * x * (1.f + erff(x * 0.70710678118f));
      if (resid) x += resid[(size_t)m * N + n];
      size_t o = (size_t)m * N + n;
      if (outF) outF[o] = x;
      if (outB) outB[o] = f2bf(x);
    }
  }
}

// ===================== instance-norm =====================
// stats[0..255]=sum(b*128+c), stats[256..511]=sumsq
__global__ void k_colstats(const float* __restrict__ x, float* __restrict__ stats){
  int tid = threadIdx.x;
  int c = tid & 127; int g = tid >> 7;
  int r0 = blockIdx.x * 256 + g * 128;
  float s = 0.f, q = 0.f;
  for (int i = 0; i < 128; ++i){
    float v = x[(size_t)(r0 + i) * 128 + c];
    s += v; q += v * v;
  }
  int b = blockIdx.x >> 7;   // 128 blocks per batch (256 rows each)
  atomicAdd(&stats[b * 128 + c], s);
  atomicAdd(&stats[256 + b * 128 + c], q);
}

__global__ void k_finstats(const float* __restrict__ stats, float* __restrict__ mr){
  int i = threadIdx.x;
  if (i >= 256) return;
  float inv = 1.f / 32768.f;
  float m = stats[i] * inv;
  float v = stats[256 + i] * inv - m * m;
  mr[i] = m;
  mr[256 + i] = rsqrtf(v + 1e-5f);
}

__global__ void k_inorm_relu(float* __restrict__ x, const float* __restrict__ mr,
                             unsigned short* __restrict__ outB){
  size_t idx = (size_t)blockIdx.x * 256 + threadIdx.x;  // < 65536*128
  int c = (int)(idx & 127); size_t t = idx >> 7; int b = (int)(t >> 15);
  float v = (x[idx] - mr[b * 128 + c]) * mr[256 + b * 128 + c];
  v = fmaxf(v, 0.f);
  x[idx] = v;
  if (outB) outB[idx] = f2bf(v);
}

__global__ void k_final(const float* __restrict__ z, const float* __restrict__ mr,
                        float* __restrict__ out){
  size_t idx = (size_t)blockIdx.x * 256 + threadIdx.x;
  int c = (int)(idx & 127); size_t t = idx >> 7; int b = (int)(t >> 15); int v = (int)(t & 32767);
  float val = fmaxf((z[idx] - mr[b * 128 + c]) * mr[256 + b * 128 + c], 0.f);
  out[((size_t)b * 128 + c) * 32768 + v] = val;   // channel-first output
}

// ===================== LayerNorm (per token row of 128) =====================
__global__ void k_ln(const float* __restrict__ x, const float* __restrict__ g,
                     const float* __restrict__ bb, unsigned short* __restrict__ out){
  int tid = threadIdx.x; int wid = tid >> 5, lane = tid & 31;
  size_t t = (size_t)blockIdx.x * 8 + wid;
  float4 v = ((const float4*)(x + t * 128))[lane];
  float s = v.x + v.y + v.z + v.w;
#pragma unroll
  for (int o = 1; o < 32; o <<= 1) s += __shfl_xor(s, o, 32);
  float mean = s * (1.f / 128.f);
  float dx = v.x - mean, dy = v.y - mean, dz = v.z - mean, dw = v.w - mean;
  float q = dx * dx + dy * dy + dz * dz + dw * dw;
#pragma unroll
  for (int o = 1; o < 32; o <<= 1) q += __shfl_xor(q, o, 32);
  float rstd = rsqrtf(q * (1.f / 128.f) + 1e-5f);
  int c = lane * 4;
  out[t * 128 + c + 0] = f2bf(dx * rstd * g[c + 0] + bb[c + 0]);
  out[t * 128 + c + 1] = f2bf(dy * rstd * g[c + 1] + bb[c + 1]);
  out[t * 128 + c + 2] = f2bf(dz * rstd * g[c + 2] + bb[c + 2]);
  out[t * 128 + c + 3] = f2bf(dw * rstd * g[c + 3] + bb[c + 3]);
}

__global__ void k_rowmean(const float* __restrict__ x, float* __restrict__ avg){
  int tid = threadIdx.x; int wid = tid >> 5, lane = tid & 31;
  size_t t = (size_t)blockIdx.x * 8 + wid;
  float4 v = ((const float4*)(x + t * 128))[lane];
  float s = v.x + v.y + v.z + v.w;
#pragma unroll
  for (int o = 1; o < 32; o <<= 1) s += __shfl_xor(s, o, 32);
  if (lane == 0) avg[t] = s * (1.f / 128.f);
}

// 1-channel 3x3x3 conv + sigmoid on avg
__global__ void k_acconv(const float* __restrict__ avg, const float* __restrict__ wc,
                         const float* __restrict__ bc, float* __restrict__ ac){
  int idx = blockIdx.x * 256 + threadIdx.x;
  if (idx >= 65536) return;
  int v = idx & 32767; int b = idx >> 15;
  int w = v & 31, h = (v >> 5) & 31, d = v >> 10;
  float s = bc[0];
#pragma unroll
  for (int t = 0; t < 27; ++t){
    int kd = t / 9 - 1, kh = (t / 3) % 3 - 1, kw = t % 3 - 1;
    int zd = d + kd, zh = h + kh, zw = w + kw;
    if ((unsigned)zd < 32u && (unsigned)zh < 32u && (unsigned)zw < 32u)
      s += wc[t] * avg[b * 32768 + ((zd * 32 + zh) * 32 + zw)];
  }
  ac[idx] = 1.f / (1.f + __expf(-s));
}

// concat(ac*y1, yst) -> bf16 [tok][256]
__global__ void k_concat(const float* __restrict__ y1, const float* __restrict__ acv,
                         const float* __restrict__ yst, unsigned short* __restrict__ out){
  size_t idx = (size_t)blockIdx.x * 256 + threadIdx.x;  // < 65536*256
  int k = (int)(idx & 255); size_t t = idx >> 8;
  float v = (k < 128) ? acv[t] * y1[t * 128 + k] : yst[t * 128 + (k - 128)];
  out[idx] = f2bf(v);
}

// ===================== windowed attention =====================
DEV int wlab(int t, int dB, int hB, int wB){
  int td = t >> 4, th = (t >> 2) & 3, tw = t & 3;
  int cd = dB * 4 + td, ch = hB * 4 + th, cw = wB * 4 + tw;
  int a = cd < 28 ? 0 : (cd < 30 ? 1 : 2);
  int b = ch < 28 ? 0 : (ch < 30 ? 1 : 2);
  int c = cw < 28 ? 0 : (cw < 30 ? 1 : 2);
  return a * 9 + b * 3 + c;
}

__global__ void __launch_bounds__(256) k_attn(const unsigned short* __restrict__ qkv,
                                              unsigned short* __restrict__ o, int shift){
  extern __shared__ unsigned short smem[];
  unsigned short* tok  = smem;               // 64 * 392
  unsigned short* pbuf = smem + 64 * 392;    // 8 waves * 64 * 72
  int tid = threadIdx.x, wid = tid >> 5, lane = tid & 31;
  int blk = blockIdx.x;
  int b = blk >> 9; int nw = blk & 511;
  int wB = nw & 7, hB = (nw >> 3) & 7, dB = nw >> 6;

  // async gather of window qkv rows (shift-rolled)
  {
    int t = tid >> 2, part = tid & 3;
    int td = t >> 4, th = (t >> 2) & 3, tw = t & 3;
    int sd = (dB * 4 + td + shift) & 31;
    int sh = (hB * 4 + th + shift) & 31;
    int sw = (wB * 4 + tw + shift) & 31;
    size_t vox = (size_t)((sd * 32 + sh) * 32 + sw);
    const unsigned short* src = qkv + ((size_t)b * 32768 + vox) * 384 + part * 96;
    unsigned short* dst = tok + t * 392 + part * 96;
#pragma unroll
    for (int i = 0; i < 12; ++i) async_ld_b128(src + i * 8, dst + i * 8);
  }
  wait_async0();
  __syncthreads();

  const unsigned short* qb = tok + wid * 16;
  const unsigned short* kb = tok + 128 + wid * 16;
  const unsigned short* vb = tok + 256 + wid * 16;

  // S = Q K^T (K padded 16->32)
  FragF S[4][4];
#pragma unroll
  for (int mi = 0; mi < 4; ++mi)
#pragma unroll
    for (int ni = 0; ni < 4; ++ni) S[mi][ni].v = vzero8();
  v16bf fa[4];
#pragma unroll
  for (int mi = 0; mi < 4; ++mi) fa[mi] = fragA_load_pad16(qb + mi * 16 * 392, 392);
#pragma unroll
  for (int ni = 0; ni < 4; ++ni){
    v16bf fb = fragB_load_kt_pad16(kb + ni * 16 * 392, 392);
#pragma unroll
    for (int mi = 0; mi < 4; ++mi) S[mi][ni].v = wmma_bf16(fa[mi], fb, S[mi][ni].v);
  }

  // softmax rows (16-lane halves hold a row across 4 N-tiles)
  int hf = lane >> 4, nl = lane & 15;
  unsigned short* pw = pbuf + wid * 64 * 72;
#pragma unroll
  for (int mi = 0; mi < 4; ++mi){
#pragma unroll
    for (int v = 0; v < 8; ++v){
      int m = mi * 16 + v + 8 * hf;
      float vals[4];
      float mx = -3.0e38f;
      int lr = shift ? wlab(m, dB, hB, wB) : 0;
#pragma unroll
      for (int ni = 0; ni < 4; ++ni){
        int n = ni * 16 + nl;
        float s = S[mi][ni].f[v] * 0.25f;
        if (shift && (wlab(n, dB, hB, wB) != lr)) s -= 1e9f;
        vals[ni] = s; mx = fmaxf(mx, s);
      }
#pragma unroll
      for (int off = 1; off < 16; off <<= 1) mx = fmaxf(mx, __shfl_xor(mx, off, 32));
      float sum = 0.f;
#pragma unroll
      for (int ni = 0; ni < 4; ++ni){ vals[ni] = __expf(vals[ni] - mx); sum += vals[ni]; }
#pragma unroll
      for (int off = 1; off < 16; off <<= 1) sum += __shfl_xor(sum, off, 32);
      float inv = 1.f / sum;
#pragma unroll
      for (int ni = 0; ni < 4; ++ni) pw[m * 72 + ni * 16 + nl] = f2bf(vals[ni] * inv);
    }
  }
  __syncthreads();

  // O = P V  (K = 64 tokens, two 32-steps)
  FragF O[4];
#pragma unroll
  for (int mi = 0; mi < 4; ++mi) O[mi].v = vzero8();
#pragma unroll
  for (int ks = 0; ks < 2; ++ks){
    v16bf fbv = fragB_load_kmajor(vb + ks * 32 * 392, 392);
#pragma unroll
    for (int mi = 0; mi < 4; ++mi){
      v16bf fap = fragA_load(pw + mi * 16 * 72 + ks * 32, 72);
      O[mi].v = wmma_bf16(fap, fbv, O[mi].v);
    }
  }
  // scatter O back to (rolled) voxel space
#pragma unroll
  for (int mi = 0; mi < 4; ++mi){
#pragma unroll
    for (int v = 0; v < 8; ++v){
      int t = mi * 16 + v + 8 * hf;
      int td = t >> 4, th = (t >> 2) & 3, tw = t & 3;
      int sd = (dB * 4 + td + shift) & 31;
      int sh2 = (hB * 4 + th + shift) & 31;
      int sw2 = (wB * 4 + tw + shift) & 31;
      size_t vox = (size_t)((sd * 32 + sh2) * 32 + sw2);
      o[((size_t)b * 32768 + vox) * 128 + wid * 16 + nl] = f2bf(O[mi].f[v]);
    }
  }
}

// ===================== host orchestration =====================
extern "C" void kernel_launch(void* const* d_in, const int* in_sizes, int n_in,
                              void* d_out, int out_size, void* d_ws, size_t ws_size,
                              hipStream_t stream){
  (void)in_sizes; (void)n_in; (void)out_size; (void)ws_size;
  const int TOK = 65536;            // B * 32^3

  const float* x    = (const float*)d_in[0];
  const float* wy   = (const float*)d_in[1];
  const float* w1c  = (const float*)d_in[3];
  const float* wc   = (const float*)d_in[5];
  const float* bc   = (const float*)d_in[6];
  const float* wz   = (const float*)d_in[7];

  char* ws = (char*)d_ws;
  size_t off = 0;
  auto alloc = [&](size_t bytes){ size_t o = off; off += (bytes + 255) & ~(size_t)255; return o; };

  unsigned short* xT     = (unsigned short*)(ws + alloc((size_t)TOK * 64 * 2));
  unsigned short* WyT    = (unsigned short*)(ws + alloc((size_t)27 * 128 * 64 * 2));
  unsigned short* W1cT   = (unsigned short*)(ws + alloc((size_t)27 * 128 * 128 * 2));
  unsigned short* WqkvT[2]; unsigned short* WpT[2]; unsigned short* Wm1T[2]; unsigned short* Wm2T[2];
  for (int i = 0; i < 2; ++i){
    WqkvT[i] = (unsigned short*)(ws + alloc((size_t)384 * 128 * 2));
    WpT[i]   = (unsigned short*)(ws + alloc((size_t)128 * 128 * 2));
    Wm1T[i]  = (unsigned short*)(ws + alloc((size_t)512 * 128 * 2));
    Wm2T[i]  = (unsigned short*)(ws + alloc((size_t)128 * 512 * 2));
  }
  unsigned short* WzT    = (unsigned short*)(ws + alloc((size_t)128 * 256 * 2));
  float*          xsw    = (float*)(ws + alloc((size_t)TOK * 128 * 4));          // y -> swin residual
  unsigned short* ybf    = (unsigned short*)(ws + alloc((size_t)TOK * 128 * 2));
  float*          y1     = (float*)(ws + alloc((size_t)TOK * 128 * 4));          // also final z_raw
  float*          avg    = (float*)(ws + alloc((size_t)TOK * 4));
  float*          acb    = (float*)(ws + alloc((size_t)TOK * 4));
  unsigned short* lnbuf  = (unsigned short*)(ws + alloc((size_t)TOK * 128 * 2)); // LN out / attn out
  unsigned short* big    = (unsigned short*)(ws + alloc((size_t)TOK * 512 * 2)); // qkv / mlp-h / concat
  float*          stats  = (float*)(ws + alloc(512 * 4));
  float*          mr     = (float*)(ws + alloc(512 * 4));

  const size_t attn_lds = (size_t)64 * 392 * 2 + (size_t)8 * 64 * 72 * 2;

  // ---- weight / input prep (bf16) ----
  k_xpose<<<16384, 256, 0, stream>>>(x, xT);
  k_wprep_conv<<<(27 * 128 * 64 + 255) / 256, 256, 0, stream>>>(wy, WyT, 64);
  k_wprep_conv<<<(27 * 128 * 128 + 255) / 256, 256, 0, stream>>>(w1c, W1cT, 128);
  for (int blk = 0; blk < 2; ++blk){
    int base = 9 + blk * 12;
    k_wprep_mat<<<(128 * 384 + 255) / 256, 256, 0, stream>>>((const float*)d_in[base + 2], WqkvT[blk], 128, 384, 1);
    k_wprep_mat<<<(128 * 128 + 255) / 256, 256, 0, stream>>>((const float*)d_in[base + 4], WpT[blk], 128, 128, 1);
    k_wprep_mat<<<(128 * 512 + 255) / 256, 256, 0, stream>>>((const float*)d_in[base + 8], Wm1T[blk], 128, 512, 1);
    k_wprep_mat<<<(512 * 128 + 255) / 256, 256, 0, stream>>>((const float*)d_in[base + 10], Wm2T[blk], 512, 128, 1);
  }
  k_wprep_mat<<<(128 * 256 + 255) / 256, 256, 0, stream>>>(wz, WzT, 256, 128, 0);

  // ---- conv1 + InstanceNorm + ReLU (bias irrelevant before affine-free IN) ----
  k_conv3<<<1024, 256, 0, stream>>>(xT, WyT, xsw, 64);
  hipMemsetAsync(stats, 0, 2048, stream);
  k_colstats<<<256, 256, 0, stream>>>(xsw, stats);
  k_finstats<<<1, 256, 0, stream>>>(stats, mr);
  k_inorm_relu<<<32768, 256, 0, stream>>>(xsw, mr, ybf);

  // ---- conv2 + IN + ReLU ----
  k_conv3<<<1024, 256, 0, stream>>>(ybf, W1cT, y1, 128);
  hipMemsetAsync(stats, 0, 2048, stream);
  k_colstats<<<256, 256, 0, stream>>>(y1, stats);
  k_finstats<<<1, 256, 0, stream>>>(stats, mr);
  k_inorm_relu<<<32768, 256, 0, stream>>>(y1, mr, nullptr);

  // ---- channel-attention branch ----
  k_rowmean<<<8192, 256, 0, stream>>>(y1, avg);
  k_acconv<<<256, 256, 0, stream>>>(avg, wc, bc, acb);

  // ---- two Swin blocks (residual stream in xsw) ----
  for (int blk = 0; blk < 2; ++blk){
    int base = 9 + blk * 12;
    const float* g1 = (const float*)d_in[base + 0];
    const float* b1 = (const float*)d_in[base + 1];
    const float* bq = (const float*)d_in[base + 3];
    const float* bp = (const float*)d_in[base + 5];
    const float* g2 = (const float*)d_in[base + 6];
    const float* b2 = (const float*)d_in[base + 7];
    const float* bm1 = (const float*)d_in[base + 9];
    const float* bm2 = (const float*)d_in[base + 11];
    int shift = blk ? 2 : 0;

    k_ln<<<8192, 256, 0, stream>>>(xsw, g1, b1, lnbuf);
    k_gemm<<<dim3(1024, 6), 256, 0, stream>>>(lnbuf, WqkvT[blk], bq, nullptr,
                                              nullptr, big, TOK, 384, 128, 0);
    k_attn<<<1024, 256, attn_lds, stream>>>(big, lnbuf, shift);
    k_gemm<<<dim3(1024, 2), 256, 0, stream>>>(lnbuf, WpT[blk], bp, xsw,
                                              xsw, nullptr, TOK, 128, 128, 0);
    k_ln<<<8192, 256, 0, stream>>>(xsw, g2, b2, lnbuf);
    k_gemm<<<dim3(1024, 8), 256, 0, stream>>>(lnbuf, Wm1T[blk], bm1, nullptr,
                                              nullptr, big, TOK, 512, 128, 1);
    k_gemm<<<dim3(1024, 2), 256, 0, stream>>>(big, Wm2T[blk], bm2, xsw,
                                              xsw, nullptr, TOK, 128, 512, 0);
  }

  // ---- concat + 1x1x1 conv + IN + ReLU -> channel-first output ----
  k_concat<<<65536, 256, 0, stream>>>(y1, acb, xsw, big);
  k_gemm<<<dim3(1024, 2), 256, 0, stream>>>(big, WzT, nullptr, nullptr,
                                            y1, nullptr, TOK, 128, 256, 0);
  hipMemsetAsync(stats, 0, 2048, stream);
  k_colstats<<<256, 256, 0, stream>>>(y1, stats);
  k_finstats<<<1, 256, 0, stream>>>(stats, mr);
  k_final<<<32768, 256, 0, stream>>>(y1, mr, (float*)d_out);
}
